// DecoderLayer_44263932953096
// MI455X (gfx1250) — matrix operand order repaired and checked by reading, not compiled, not run on hardware
//
#include <hip/hip_runtime.h>
#include <hip/hip_bf16.h>
#include <math.h>

typedef __bf16 bf16;
typedef __attribute__((ext_vector_type(16))) __bf16 v16bf;
typedef __attribute__((ext_vector_type(8)))  __bf16 v8bf;
typedef __attribute__((ext_vector_type(8)))  float  v8f;

#define S_LEN  2048
#define DMODEL 1024
#define NHEAD  16
#define HDIM   64
#define NEXP   8
#define FFH    4096
#define QKVN   (NHEAD*HDIM + 2*HDIM)   // 1152
#define SEGAL  32                       // expert segment alignment (rows)
#define MAXG   (S_LEN*2 + NEXP*SEGAL)   // 4352 gathered rows max
#define MAXT32 (MAXG/32)                // 136 32-row tiles
#define EPSV   1e-5f

union BF16x16 { v16bf v; v8bf h[2]; };

// ---- WMMA fragment helpers (layouts per CDNA5 ISA 7.12.2) ----
// A 16x32 bf16: lane l -> m = l&15, kh = l>>4; elems 0..7 = K k0+kh*8.., 8..15 = K k0+16+kh*8..
__device__ __forceinline__ v16bf load_a_frag(const bf16* base, size_t row, int ld, int k0, int kh) {
  const bf16* p = base + row * (size_t)ld + (size_t)(k0 + kh * 8);
  BF16x16 r;
  r.h[0] = *reinterpret_cast<const v8bf*>(p);
  r.h[1] = *reinterpret_cast<const v8bf*>(p + 16);
  return r.v;
}
// B 32x16 bf16 (as Bt[N][K]): lane l -> n = l&15, kh = l>>4; elems 0..7 = K k0+kh*8.., 8..15 = K k0+16+kh*8..
__device__ __forceinline__ v16bf load_b_frag(const bf16* base, size_t row, int ld, int k0, int kh) {
  const bf16* p = base + row * (size_t)ld + (size_t)(k0 + kh * 8);
  BF16x16 r;
  r.h[0] = *reinterpret_cast<const v8bf*>(p);
  r.h[1] = *reinterpret_cast<const v8bf*>(p + 16);
  return r.v;
}
__device__ __forceinline__ v8f wmma_bf16(v16bf a, v16bf b, v8f c) {
  return __builtin_amdgcn_wmma_f32_16x16x32_bf16(false, a, false, b, (short)0, c, false, false);
}

__device__ __forceinline__ float half16_max(float v) {
  v = fmaxf(v, __shfl_xor(v, 1, 32));
  v = fmaxf(v, __shfl_xor(v, 2, 32));
  v = fmaxf(v, __shfl_xor(v, 4, 32));
  v = fmaxf(v, __shfl_xor(v, 8, 32));
  return v;
}
__device__ __forceinline__ float half16_sum(float v) {
  v += __shfl_xor(v, 1, 32);
  v += __shfl_xor(v, 2, 32);
  v += __shfl_xor(v, 4, 32);
  v += __shfl_xor(v, 8, 32);
  return v;
}
__device__ __forceinline__ float block_reduce_sum(float v, float* red) {
  int tid = threadIdx.x;
  __syncthreads();
  red[tid] = v;
  __syncthreads();
  for (int st = 128; st > 0; st >>= 1) {
    if (tid < st) red[tid] += red[tid + st];
    __syncthreads();
  }
  float r = red[0];
  __syncthreads();
  return r;
}
__device__ __forceinline__ float gelu_tanh(float x) {
  float x3 = x * x * x;
  return 0.5f * x * (1.0f + tanhf(0.7978845608028654f * (x + 0.044715f * x3)));
}

// ---------------- weight convert + transpose: dst[c*R + r] = bf16(src[r*C + c]) ----------------
__global__ void k_convT(const float* __restrict__ src, bf16* __restrict__ dst,
                        int R, int C, size_t sbatch, size_t dbatch) {
  size_t n = (size_t)R * C;
  size_t i = (size_t)blockIdx.x * blockDim.x + threadIdx.x;
  if (i >= n) return;
  const float* s = src + (size_t)blockIdx.y * sbatch;
  bf16* d = dst + (size_t)blockIdx.y * dbatch;
  size_t c = i / (size_t)R, r = i % (size_t)R;
  d[c * (size_t)R + r] = (bf16)s[r * (size_t)C + c];
}

// ---------------- rmsnorm: h1b = bf16(rmsnorm(x, g)) ----------------
__global__ void k_rms_pre(const float* __restrict__ x, const float* __restrict__ g,
                          bf16* __restrict__ out) {
  __shared__ float red[256];
  int row = blockIdx.x, tid = threadIdx.x;
  const float* xr = x + (size_t)row * DMODEL;
  float s = 0.f;
  float v[4];
  for (int i = 0; i < 4; ++i) { v[i] = xr[tid + i * 256]; s += v[i] * v[i]; }
  float tot = block_reduce_sum(s, red);
  float inv = rsqrtf(tot / (float)DMODEL + EPSV);
  for (int i = 0; i < 4; ++i) {
    int d = tid + i * 256;
    out[(size_t)row * DMODEL + d] = (bf16)(v[i] * inv * g[d]);
  }
}

// ------- generic GEMM: C[M,N]f32 = A[M,K]bf16 * Bt[N,K]bf16; 32 M-rows x 64 N-cols per wave -------
__global__ void k_gemm_nt(const bf16* __restrict__ A, int lda,
                          const bf16* __restrict__ Bt, int ldb,
                          float* __restrict__ C, int ldc,
                          int N, int Kd) {
  int wave = threadIdx.x >> 5, lane = threadIdx.x & 31;
  int m0 = blockIdx.x * 32;
  int n0 = (blockIdx.y * 8 + wave) * 64;
  if (n0 >= N) return;
  int ln = lane & 15, kh = lane >> 4;
  v8f zero = {};
  v8f acc0[4] = {zero, zero, zero, zero};
  v8f acc1[4] = {zero, zero, zero, zero};
  for (int k0 = 0; k0 < Kd; k0 += 32) {
    // prefetch weight stream ~2 K-steps ahead (global_prefetch_b8; speculative, OOB safe)
    __builtin_prefetch(Bt + (size_t)(n0 + ln) * ldb + k0 + 64, 0, 1);
    v16bf a0 = load_a_frag(A, (size_t)(m0 + ln), lda, k0, kh);
    v16bf a1 = load_a_frag(A, (size_t)(m0 + 16 + ln), lda, k0, kh);
#pragma unroll
    for (int t = 0; t < 4; ++t) {
      v16bf b = load_b_frag(Bt, (size_t)(n0 + t * 16 + ln), ldb, k0, kh);
      acc0[t] = wmma_bf16(a0, b, acc0[t]);
      acc1[t] = wmma_bf16(a1, b, acc1[t]);
    }
  }
#pragma unroll
  for (int t = 0; t < 4; ++t)
#pragma unroll
    for (int r = 0; r < 8; ++r) {
      C[(size_t)(m0 + r + 8 * kh) * ldc + (n0 + t * 16 + ln)] = acc0[t][r];
      C[(size_t)(m0 + 16 + r + 8 * kh) * ldc + (n0 + t * 16 + ln)] = acc1[t][r];
    }
}

// ---------------- split qkv f32 -> qb/kb/vT bf16 ----------------
__global__ void k_pack_qkv(const float* __restrict__ qkv, bf16* __restrict__ qb,
                           bf16* __restrict__ kb, bf16* __restrict__ vTb) {
  size_t i = (size_t)blockIdx.x * blockDim.x + threadIdx.x;
  if (i >= (size_t)S_LEN * QKVN) return;
  int s = (int)(i / QKVN), c = (int)(i % QKVN);
  bf16 v = (bf16)qkv[i];
  if (c < NHEAD * HDIM)            qb[(size_t)s * DMODEL + c] = v;
  else if (c < NHEAD * HDIM + HDIM) kb[(size_t)s * HDIM + (c - NHEAD * HDIM)] = v;
  else                              vTb[(size_t)(c - NHEAD * HDIM - HDIM) * S_LEN + s] = v;
}

// ---------------- flash attention: one wave per (head, 16-query tile) ----------------
__global__ void k_flash(const bf16* __restrict__ qb, const bf16* __restrict__ kb,
                        const bf16* __restrict__ vTb, bf16* __restrict__ attnb) {
  __shared__ __align__(16) bf16 pshare[8][16 * 32];
  int wave = threadIdx.x >> 5, lane = threadIdx.x & 31;
  int h = blockIdx.x;
  int q0 = (blockIdx.y * 8 + wave) * 16;
  int ln = lane & 15, kh = lane >> 4;
  v16bf aq0 = load_a_frag(qb, (size_t)(q0 + ln), DMODEL, h * HDIM + 0, kh);
  v16bf aq1 = load_a_frag(qb, (size_t)(q0 + ln), DMODEL, h * HDIM + 32, kh);
  v8f zero = {};
  v8f o[4] = {zero, zero, zero, zero};
  float rM[8], rL[8];
#pragma unroll
  for (int r = 0; r < 8; ++r) { rM[r] = -1e30f; rL[r] = 0.f; }

  for (int j = 0; j < S_LEN; j += 32) {
    v8f s0 = {}, s1 = {};
    {
      v16bf b = load_b_frag(kb, (size_t)(j + ln), HDIM, 0, kh);
      s0 = wmma_bf16(aq0, b, s0);
      b = load_b_frag(kb, (size_t)(j + ln), HDIM, 32, kh);
      s0 = wmma_bf16(aq1, b, s0);
      b = load_b_frag(kb, (size_t)(j + 16 + ln), HDIM, 0, kh);
      s1 = wmma_bf16(aq0, b, s1);
      b = load_b_frag(kb, (size_t)(j + 16 + ln), HDIM, 32, kh);
      s1 = wmma_bf16(aq1, b, s1);
    }
#pragma unroll
    for (int r = 0; r < 8; ++r) {
      float x0 = s0[r] * 0.125f, x1v = s1[r] * 0.125f;
      float mx = half16_max(fmaxf(x0, x1v));
      float nM = fmaxf(rM[r], mx);
      float corr = __expf(rM[r] - nM);
      rM[r] = nM;
      float p0 = __expf(x0 - nM), p1 = __expf(x1v - nM);
      float rs = half16_sum(p0 + p1);
      rL[r] = rL[r] * corr + rs;
#pragma unroll
      for (int t = 0; t < 4; ++t) o[t][r] *= corr;
      int prow = r + 8 * kh;
      pshare[wave][prow * 32 + ln] = (bf16)p0;
      pshare[wave][prow * 32 + 16 + ln] = (bf16)p1;
    }
    // re-read P in A-fragment layout (C layout -> A layout transpose through LDS)
    BF16x16 pa;
    pa.h[0] = *reinterpret_cast<const v8bf*>(&pshare[wave][ln * 32 + kh * 8]);
    pa.h[1] = *reinterpret_cast<const v8bf*>(&pshare[wave][ln * 32 + 16 + kh * 8]);
#pragma unroll
    for (int t = 0; t < 4; ++t) {
      v16bf bv = load_b_frag(vTb, (size_t)(t * 16 + ln), S_LEN, j, kh);
      o[t] = wmma_bf16(pa.v, bv, o[t]);
    }
  }
#pragma unroll
  for (int t = 0; t < 4; ++t)
#pragma unroll
    for (int r = 0; r < 8; ++r) {
      float outv = o[t][r] / rL[r];
      attnb[(size_t)(q0 + r + 8 * kh) * DMODEL + h * HDIM + t * 16 + ln] = (bf16)outv;
    }
}

// ---------------- residual + rmsnorm (post-attn) + pre-moe rmsnorm ----------------
__global__ void k_post_attn(const float* __restrict__ x, const float* __restrict__ attno,
                            const float* __restrict__ gpost, const float* __restrict__ gpre,
                            float* __restrict__ x1, float* __restrict__ h2f,
                            bf16* __restrict__ h2b) {
  __shared__ float red[256];
  int row = blockIdx.x, tid = threadIdx.x;
  const float* ar = attno + (size_t)row * DMODEL;
  const float* xr = x + (size_t)row * DMODEL;
  float a[4];
  float s = 0.f;
  for (int i = 0; i < 4; ++i) { a[i] = ar[tid + i * 256]; s += a[i] * a[i]; }
  float inv1 = rsqrtf(block_reduce_sum(s, red) / (float)DMODEL + EPSV);
  float v[4];
  float s2 = 0.f;
  for (int i = 0; i < 4; ++i) {
    int d = tid + i * 256;
    v[i] = xr[d] + a[i] * inv1 * gpost[d];
    x1[(size_t)row * DMODEL + d] = v[i];
    s2 += v[i] * v[i];
  }
  float inv2 = rsqrtf(block_reduce_sum(s2, red) / (float)DMODEL + EPSV);
  for (int i = 0; i < 4; ++i) {
    int d = tid + i * 256;
    float hh = v[i] * inv2 * gpre[d];
    h2f[(size_t)row * DMODEL + d] = hh;
    h2b[(size_t)row * DMODEL + d] = (bf16)hh;
  }
}

// ---------------- router: logits, softmax probs (to d_out), top-2 ----------------
__global__ void k_router(const float* __restrict__ h2f, const float* __restrict__ Wr,
                         float* __restrict__ probs_out, int* __restrict__ top_idx,
                         float* __restrict__ top_w) {
  int wave = threadIdx.x >> 5, lane = threadIdx.x & 31;
  int t = blockIdx.x * 8 + wave;
  const float* hr = h2f + (size_t)t * DMODEL;
  float logit[NEXP];
#pragma unroll
  for (int e = 0; e < NEXP; ++e) {
    float s = 0.f;
    for (int d = lane; d < DMODEL; d += 32) s += hr[d] * Wr[(size_t)d * NEXP + e];
    s += __shfl_xor(s, 1, 32);  s += __shfl_xor(s, 2, 32);
    s += __shfl_xor(s, 4, 32);  s += __shfl_xor(s, 8, 32);
    s += __shfl_xor(s, 16, 32);
    logit[e] = s;
  }
  if (lane == 0) {
    float mx = logit[0];
    for (int e = 1; e < NEXP; ++e) mx = fmaxf(mx, logit[e]);
    float p[NEXP], denom = 0.f;
    for (int e = 0; e < NEXP; ++e) { p[e] = __expf(logit[e] - mx); denom += p[e]; }
    for (int e = 0; e < NEXP; ++e) {
      p[e] /= denom;
      probs_out[(size_t)t * NEXP + e] = p[e];
    }
    int i0 = 0;
    for (int e = 1; e < NEXP; ++e) if (p[e] > p[i0]) i0 = e;
    int i1 = (i0 == 0) ? 1 : 0;
    for (int e = 0; e < NEXP; ++e) if (e != i0 && p[e] > p[i1]) i1 = e;
    top_idx[t * 2 + 0] = i0;  top_idx[t * 2 + 1] = i1;
    top_w[t * 2 + 0] = p[i0]; top_w[t * 2 + 1] = p[i1];
  }
}

// ------- build per-expert gathered routing; segments aligned to 32 rows (deterministic) -------
__global__ void k_build_routing(const int* __restrict__ top_idx, int* __restrict__ rowmap,
                                int* __restrict__ slot2row, int* __restrict__ tileExpert) {
  if (threadIdx.x != 0 || blockIdx.x != 0) return;
  int cnt[NEXP];
  for (int e = 0; e < NEXP; ++e) cnt[e] = 0;
  for (int i = 0; i < S_LEN * 2; ++i) cnt[top_idx[i]]++;
  int off[NEXP]; int cur = 0;
  for (int e = 0; e < NEXP; ++e) { off[e] = cur; cur += (cnt[e] + SEGAL - 1) & ~(SEGAL - 1); }
  for (int g = 0; g < MAXG; ++g) rowmap[g] = 0;  // padded rows read token 0 (output unused)
  int fill[NEXP];
  for (int e = 0; e < NEXP; ++e) fill[e] = 0;
  for (int t = 0; t < S_LEN; ++t)
    for (int s = 0; s < 2; ++s) {
      int e = top_idx[t * 2 + s];
      int g = off[e] + fill[e]++;
      rowmap[g] = t * 2 + s;
      slot2row[t * 2 + s] = g;
    }
  for (int tile = 0; tile < MAXT32; ++tile) tileExpert[tile] = -1;
  for (int e = 0; e < NEXP; ++e) {
    int a = (cnt[e] + SEGAL - 1) / SEGAL;
    int base = off[e] / SEGAL;
    for (int i = 0; i < a; ++i) tileExpert[base + i] = e;
  }
}

// ------- MoE GEMM1: hid[g,FH] = bf16(gelu(h2b[rowmap] @ W1T_e)); 32 rows x 64 cols / wave -------
__global__ void k_moe_gemm1(const bf16* __restrict__ h2b, const bf16* __restrict__ W1T,
                            const int* __restrict__ rowmap, const int* __restrict__ tileExpert,
                            bf16* __restrict__ hid) {
  int e = tileExpert[blockIdx.x];
  if (e < 0) return;
  int wave = threadIdx.x >> 5, lane = threadIdx.x & 31;
  int g0 = blockIdx.x * 32;
  int n0 = (blockIdx.y * 8 + wave) * 64;
  int ln = lane & 15, kh = lane >> 4;
  int tok0 = rowmap[g0 + ln] >> 1;
  int tok1 = rowmap[g0 + 16 + ln] >> 1;
  const bf16* A0 = h2b + (size_t)tok0 * DMODEL;
  const bf16* A1 = h2b + (size_t)tok1 * DMODEL;
  const bf16* Bt = W1T + (size_t)e * FFH * DMODEL;
  v8f zero = {};
  v8f acc0[4] = {zero, zero, zero, zero};
  v8f acc1[4] = {zero, zero, zero, zero};
  for (int k0 = 0; k0 < DMODEL; k0 += 32) {
    __builtin_prefetch(Bt + (size_t)(n0 + ln) * DMODEL + k0 + 64, 0, 1);
    BF16x16 a0, a1;
    const bf16* p0 = A0 + k0 + kh * 8;
    const bf16* p1 = A1 + k0 + kh * 8;
    a0.h[0] = *reinterpret_cast<const v8bf*>(p0);
    a0.h[1] = *reinterpret_cast<const v8bf*>(p0 + 16);
    a1.h[0] = *reinterpret_cast<const v8bf*>(p1);
    a1.h[1] = *reinterpret_cast<const v8bf*>(p1 + 16);
#pragma unroll
    for (int t = 0; t < 4; ++t) {
      v16bf b = load_b_frag(Bt, (size_t)(n0 + t * 16 + ln), DMODEL, k0, kh);
      acc0[t] = wmma_bf16(a0.v, b, acc0[t]);
      acc1[t] = wmma_bf16(a1.v, b, acc1[t]);
    }
  }
#pragma unroll
  for (int t = 0; t < 4; ++t)
#pragma unroll
    for (int r = 0; r < 8; ++r) {
      hid[(size_t)(g0 + r + 8 * kh) * FFH + (n0 + t * 16 + ln)] = (bf16)gelu_tanh(acc0[t][r]);
      hid[(size_t)(g0 + 16 + r + 8 * kh) * FFH + (n0 + t * 16 + ln)] = (bf16)gelu_tanh(acc1[t][r]);
    }
}

// ------- MoE GEMM2: eout[g,D] = hid[g] @ W2T_e; 32 rows x 64 cols / wave -------
__global__ void k_moe_gemm2(const bf16* __restrict__ hid, const bf16* __restrict__ W2T,
                            const int* __restrict__ tileExpert, float* __restrict__ eout) {
  int e = tileExpert[blockIdx.x];
  if (e < 0) return;
  int wave = threadIdx.x >> 5, lane = threadIdx.x & 31;
  int g0 = blockIdx.x * 32;
  int n0 = (blockIdx.y * 8 + wave) * 64;
  int ln = lane & 15, kh = lane >> 4;
  const bf16* Bt = W2T + (size_t)e * DMODEL * FFH;
  v8f zero = {};
  v8f acc0[4] = {zero, zero, zero, zero};
  v8f acc1[4] = {zero, zero, zero, zero};
  for (int k0 = 0; k0 < FFH; k0 += 32) {
    __builtin_prefetch(Bt + (size_t)(n0 + ln) * FFH + k0 + 64, 0, 1);
    v16bf a0 = load_a_frag(hid, (size_t)(g0 + ln), FFH, k0, kh);
    v16bf a1 = load_a_frag(hid, (size_t)(g0 + 16 + ln), FFH, k0, kh);
#pragma unroll
    for (int t = 0; t < 4; ++t) {
      v16bf b = load_b_frag(Bt, (size_t)(n0 + t * 16 + ln), FFH, k0, kh);
      acc0[t] = wmma_bf16(a0, b, acc0[t]);
      acc1[t] = wmma_bf16(a1, b, acc1[t]);
    }
  }
#pragma unroll
  for (int t = 0; t < 4; ++t)
#pragma unroll
    for (int r = 0; r < 8; ++r) {
      eout[(size_t)(g0 + r + 8 * kh) * DMODEL + (n0 + t * 16 + ln)] = acc0[t][r];
      eout[(size_t)(g0 + 16 + r + 8 * kh) * DMODEL + (n0 + t * 16 + ln)] = acc1[t][r];
    }
}

// ---------------- final: weighted combine + rmsnorm + residual -> d_out ----------------
__global__ void k_combine(const float* __restrict__ x1, const float* __restrict__ eout,
                          const int* __restrict__ slot2row, const float* __restrict__ top_w,
                          const float* __restrict__ gpost, float* __restrict__ out) {
  __shared__ float red[256];
  int t = blockIdx.x, tid = threadIdx.x;
  int g0 = slot2row[t * 2], g1 = slot2row[t * 2 + 1];
  float w0 = top_w[t * 2], w1 = top_w[t * 2 + 1];
  float v[4];
  float s = 0.f;
  for (int i = 0; i < 4; ++i) {
    int d = tid + i * 256;
    float m = w0 * eout[(size_t)g0 * DMODEL + d] + w1 * eout[(size_t)g1 * DMODEL + d];
    v[i] = m; s += m * m;
  }
  float inv = rsqrtf(block_reduce_sum(s, red) / (float)DMODEL + EPSV);
  for (int i = 0; i < 4; ++i) {
    int d = tid + i * 256;
    out[(size_t)t * DMODEL + d] = x1[(size_t)t * DMODEL + d] + v[i] * inv * gpost[d];
  }
}

// =====================================================================================
extern "C" void kernel_launch(void* const* d_in, const int* in_sizes, int n_in,
                              void* d_out, int out_size, void* d_ws, size_t ws_size,
                              hipStream_t stream) {
  (void)in_sizes; (void)n_in; (void)out_size; (void)ws_size;
  const float* x    = (const float*)d_in[0];
  const float* Wq   = (const float*)d_in[1];
  const float* Wk   = (const float*)d_in[2];
  const float* Wv   = (const float*)d_in[3];
  const float* Wo   = (const float*)d_in[4];
  const float* g_pre_mqa  = (const float*)d_in[5];
  const float* g_post_mqa = (const float*)d_in[6];
  const float* g_pre_moe  = (const float*)d_in[7];
  const float* g_post_moe = (const float*)d_in[8];
  const float* Wr   = (const float*)d_in[9];
  const float* W1   = (const float*)d_in[10];
  const float* W2   = (const float*)d_in[11];
  float* out = (float*)d_out;
  float* probs_out = out + (size_t)S_LEN * DMODEL;

  char* w = (char*)d_ws;
  size_t off = 0;
  auto alloc = [&](size_t bytes) -> void* {
    void* p = w + off;
    off = (off + bytes + 255) & ~(size_t)255;
    return p;
  };
  bf16* WqkvT = (bf16*)alloc((size_t)QKVN * DMODEL * 2);
  bf16* WoT   = (bf16*)alloc((size_t)DMODEL * DMODEL * 2);
  bf16* W1T   = (bf16*)alloc((size_t)NEXP * FFH * DMODEL * 2);
  bf16* W2T   = (bf16*)alloc((size_t)NEXP * DMODEL * FFH * 2);
  bf16* h1b   = (bf16*)alloc((size_t)S_LEN * DMODEL * 2);
  float* qkv  = (float*)alloc((size_t)S_LEN * QKVN * 4);
  bf16* qb    = (bf16*)alloc((size_t)S_LEN * DMODEL * 2);
  bf16* kb    = (bf16*)alloc((size_t)S_LEN * HDIM * 2);
  bf16* vTb   = (bf16*)alloc((size_t)HDIM * S_LEN * 2);
  bf16* attnb = (bf16*)alloc((size_t)S_LEN * DMODEL * 2);
  float* attno = (float*)alloc((size_t)S_LEN * DMODEL * 4);
  float* x1    = (float*)alloc((size_t)S_LEN * DMODEL * 4);
  float* h2f   = (float*)alloc((size_t)S_LEN * DMODEL * 4);
  bf16* h2b    = (bf16*)alloc((size_t)S_LEN * DMODEL * 2);
  int*  top_idx = (int*)alloc((size_t)S_LEN * 2 * 4);
  float* top_w  = (float*)alloc((size_t)S_LEN * 2 * 4);
  int*  rowmap  = (int*)alloc((size_t)MAXG * 4);
  int*  slot2row = (int*)alloc((size_t)S_LEN * 2 * 4);
  int*  tileExpert = (int*)alloc((size_t)MAXT32 * 4);
  bf16* hid   = (bf16*)alloc((size_t)MAXG * FFH * 2);
  float* eout = (float*)alloc((size_t)MAXG * DMODEL * 4);

  // 1) weight convert+transpose to bf16
  k_convT<<<dim3((DMODEL * (NHEAD * HDIM)) / 256, 1), 256, 0, stream>>>(
      Wq, WqkvT, DMODEL, NHEAD * HDIM, 0, 0);
  k_convT<<<dim3((DMODEL * HDIM) / 256, 1), 256, 0, stream>>>(
      Wk, WqkvT + (size_t)(NHEAD * HDIM) * DMODEL, DMODEL, HDIM, 0, 0);
  k_convT<<<dim3((DMODEL * HDIM) / 256, 1), 256, 0, stream>>>(
      Wv, WqkvT + (size_t)(NHEAD * HDIM + HDIM) * DMODEL, DMODEL, HDIM, 0, 0);
  k_convT<<<dim3((DMODEL * DMODEL) / 256, 1), 256, 0, stream>>>(
      Wo, WoT, DMODEL, DMODEL, 0, 0);
  k_convT<<<dim3(((size_t)DMODEL * FFH) / 256, NEXP), 256, 0, stream>>>(
      W1, W1T, DMODEL, FFH, (size_t)DMODEL * FFH, (size_t)FFH * DMODEL);
  k_convT<<<dim3(((size_t)FFH * DMODEL) / 256, NEXP), 256, 0, stream>>>(
      W2, W2T, FFH, DMODEL, (size_t)FFH * DMODEL, (size_t)DMODEL * FFH);

  // 2) pre-MQA rmsnorm
  k_rms_pre<<<S_LEN, 256, 0, stream>>>(x, g_pre_mqa, h1b);

  // 3) fused QKV GEMM: [2048,1024] x [1152,1024]^T
  k_gemm_nt<<<dim3(S_LEN / 32, 3), 256, 0, stream>>>(h1b, DMODEL, WqkvT, DMODEL,
                                                     qkv, QKVN, QKVN, DMODEL);
  // 4) split to bf16 q/k/vT
  k_pack_qkv<<<((size_t)S_LEN * QKVN) / 256, 256, 0, stream>>>(qkv, qb, kb, vTb);

  // 5) flash attention (wave32 WMMA, online softmax)
  k_flash<<<dim3(NHEAD, S_LEN / 128), 256, 0, stream>>>(qb, kb, vTb, attnb);

  // 6) output projection
  k_gemm_nt<<<dim3(S_LEN / 32, 2), 256, 0, stream>>>(attnb, DMODEL, WoT, DMODEL,
                                                     attno, DMODEL, DMODEL, DMODEL);
  // 7) residual + post-attn rms + pre-moe rms
  k_post_attn<<<S_LEN, 256, 0, stream>>>(x, attno, g_post_mqa, g_pre_moe, x1, h2f, h2b);

  // 8) router (writes routing_probs into second output region)
  k_router<<<S_LEN / 8, 256, 0, stream>>>(h2f, Wr, probs_out, top_idx, top_w);

  // 9) deterministic gather lists / tile->expert map (32-row aligned segments)
  k_build_routing<<<1, 64, 0, stream>>>(top_idx, rowmap, slot2row, tileExpert);

  // 10) gathered expert GEMMs (top-2 sparse: ~4x less work than dense)
  k_moe_gemm1<<<dim3(MAXT32, FFH / 512), 256, 0, stream>>>(h2b, W1T, rowmap, tileExpert, hid);
  k_moe_gemm2<<<dim3(MAXT32, DMODEL / 512), 256, 0, stream>>>(hid, W2T, tileExpert, eout);

  // 11) weighted combine + rmsnorm + residual -> d_out
  k_combine<<<S_LEN, 256, 0, stream>>>(x1, eout, slot2row, top_w, g_post_moe, out);
}
